// BaseSSMLayer_36266703848015
// MI455X (gfx1250) — compile-verified, matrix operand order checked by reading, not run on hardware
//
#include <hip/hip_runtime.h>
#include <hip/hip_bf16.h>
#include <stddef.h>
#include <stdint.h>

// ---------------------------------------------------------------------------
// Diagonal SSM layer on MI455X (gfx1250, wave32, WMMA, async LDS staging).
//   xs_bf = bf16(xs); B = xs_bf @ w_in^T   (bf16 WMMA, f32 acc)
//   h_t = lam*h_{t-1}+B_t                  (3-pass chunked scan, f32)
//   Y = H_bf @ c_out^T + d*xs              (bf16 WMMA, fused epilogue)
// GEMM: 256x128x32 block tile, 8 waves (4x2), 64x64 per wave (16 WMMA tiles),
// double-buffered LDS fed by GLOBAL_LOAD_ASYNC_TO_LDS_B128 (ASYNCcnt).
// ---------------------------------------------------------------------------

typedef __attribute__((ext_vector_type(4)))  int    v4i;
typedef __attribute__((ext_vector_type(4)))  __bf16 v4bf;
typedef __attribute__((ext_vector_type(8)))  __bf16 v8bf;
typedef __attribute__((ext_vector_type(16))) __bf16 v16bf;
typedef __attribute__((ext_vector_type(4)))  float  v4f;
typedef __attribute__((ext_vector_type(8)))  float  v8f;

#define SEQ_LEN   16384
#define DIM       2048
#define BM        256
#define BN        128
#define BK        32
#define LDS_ST    40        // bf16 elems per LDS row (80 B = 64 B data + 16 B pad)
#define CHUNK     256
#define NCHUNK    (SEQ_LEN / CHUNK)

#if defined(__AMDGCN__) && __has_builtin(__builtin_amdgcn_global_load_async_to_lds_b128)
#define USE_ASYNC_LDS 1
#else
#define USE_ASYNC_LDS 0
#endif

#define GLOBAL_AS __attribute__((address_space(1)))
#define LDS_AS    __attribute__((address_space(3)))

__device__ __forceinline__ void async_copy_b128(const void* g, void* l) {
#if USE_ASYNC_LDS
    // AS1 numeric value == flat address; AS3 numeric value == low 32 bits of
    // the generic LDS pointer, so integer casts build the typed pointers.
    __builtin_amdgcn_global_load_async_to_lds_b128(
        (GLOBAL_AS v4i*)(uintptr_t)g,
        (LDS_AS v4i*)(uint32_t)(uintptr_t)l,
        0, 0);
#else
    *(v8bf*)l = *(const v8bf*)g;
#endif
}

template <int N>
__device__ __forceinline__ void wait_async() {
#if USE_ASYNC_LDS
#if __has_builtin(__builtin_amdgcn_s_wait_asynccnt)
    __builtin_amdgcn_s_wait_asynccnt(N);
#else
    asm volatile("s_wait_asynccnt %0" ::"i"(N) : "memory");
#endif
#endif
}

// --------------------------- f32 -> bf16 convert ---------------------------
__global__ __launch_bounds__(256) void cvt_f32_to_bf16(const float* __restrict__ in,
                                                       __bf16* __restrict__ out,
                                                       size_t n4) {
    size_t i = (size_t)blockIdx.x * blockDim.x + threadIdx.x;
    if (i >= n4) return;
    v4f f = *(const v4f*)(in + i * 4);
    v4bf h;
    h[0] = (__bf16)f[0]; h[1] = (__bf16)f[1]; h[2] = (__bf16)f[2]; h[3] = (__bf16)f[3];
    *(v4bf*)(out + i * 4) = h;
}

// ------------------------------- WMMA GEMM ---------------------------------
// C[m,n] = sum_k A[m,k] * W[n,k]  (NT, both K-major, bf16), f32 accumulate.
template <bool EPI>
__global__ __launch_bounds__(256) void gemm_wmma_bf16(
    const __bf16* __restrict__ A, const __bf16* __restrict__ W,
    float* __restrict__ Out,
    const float* __restrict__ xs, const float* __restrict__ dskip) {
    __shared__ alignas(16) __bf16 ldsA[2][BM * LDS_ST];   // 2 x 20 KiB
    __shared__ alignas(16) __bf16 ldsB[2][BN * LDS_ST];   // 2 x 10 KiB

    const int tid   = threadIdx.x;
    const int lane  = tid & 31;
    const int wave  = tid >> 5;
    const int waveM = wave & 3;       // 4 waves along M (64 rows each)
    const int waveN = wave >> 2;      // 2 waves along N (64 cols each)
    const int m0    = blockIdx.y * BM;
    const int n0    = blockIdx.x * BN;
    const int r16   = lane & 15;
    const int g     = lane >> 4;

    const v8f vzero = {0.f, 0.f, 0.f, 0.f, 0.f, 0.f, 0.f, 0.f};
    v8f acc[4][4];
    #pragma unroll
    for (int tm = 0; tm < 4; ++tm)
        #pragma unroll
        for (int tn = 0; tn < 4; ++tn) acc[tm][tn] = vzero;

    // stage one 256x32 A tile + 128x32 W tile into LDS buffer `buf`
    // (6 async b128 per thread per K-step)
    auto stage = [&](int buf, int k0) {
        #pragma unroll
        for (int i = 0; i < 4; ++i) {         // A: 1024 chunks of 8 bf16
            int cid = tid + i * 256;
            int row = cid >> 2;
            int c8  = cid & 3;
            async_copy_b128(A + (size_t)(m0 + row) * DIM + k0 + c8 * 8,
                            &ldsA[buf][row * LDS_ST + c8 * 8]);
        }
        #pragma unroll
        for (int i = 0; i < 2; ++i) {         // W: 512 chunks of 8 bf16
            int cid = tid + i * 256;
            int row = cid >> 2;
            int c8  = cid & 3;
            async_copy_b128(W + (size_t)(n0 + row) * DIM + k0 + c8 * 8,
                            &ldsB[buf][row * LDS_ST + c8 * 8]);
        }
    };

    const int NK = DIM / BK;              // 64 K-steps
    stage(0, 0);
    for (int k = 0; k < NK; ++k) {
        const int cur = k & 1;
        if (k + 1 < NK) {
            stage(1 - cur, (k + 1) * BK); // prefetch next tile (async)
            wait_async<6>();              // tile k done, tile k+1 in flight
        } else {
            wait_async<0>();
        }
        __syncthreads();

        // fragments (ISA 16-bit 16x32 layout: lane -> row lane%16;
        // K runs [g*8, g*8+8) and [16+g*8, 16+g*8+8) per lane group g)
        v16bf afrag[4], bfrag[4];
        #pragma unroll
        for (int tm = 0; tm < 4; ++tm) {
            const __bf16* base = &ldsA[cur][(waveM * 64 + tm * 16 + r16) * LDS_ST + g * 8];
            v8bf lo = *(const v8bf*)base;
            v8bf hi = *(const v8bf*)(base + 16);
            afrag[tm] = __builtin_shufflevector(lo, hi, 0, 1, 2, 3, 4, 5, 6, 7,
                                                8, 9, 10, 11, 12, 13, 14, 15);
        }
        #pragma unroll
        for (int tn = 0; tn < 4; ++tn) {
            const __bf16* base = &ldsB[cur][(waveN * 64 + tn * 16 + r16) * LDS_ST + g * 8];
            v8bf lo = *(const v8bf*)base;
            v8bf hi = *(const v8bf*)(base + 16);
            bfrag[tn] = __builtin_shufflevector(lo, hi, 0, 1, 2, 3, 4, 5, 6, 7,
                                                8, 9, 10, 11, 12, 13, 14, 15);
        }
        #pragma unroll
        for (int tm = 0; tm < 4; ++tm)
            #pragma unroll
            for (int tn = 0; tn < 4; ++tn)
                acc[tm][tn] = __builtin_amdgcn_wmma_f32_16x16x32_bf16(
                    false, afrag[tm], false, bfrag[tn], (short)0, acc[tm][tn],
                    false, false);
        __syncthreads();   // protect buf[cur] before it is overwritten
    }

    // store: vgpr v -> M = v + 8*(lane>>4); N = lane%16
    #pragma unroll
    for (int tm = 0; tm < 4; ++tm)
        #pragma unroll
        for (int tn = 0; tn < 4; ++tn)
            #pragma unroll
            for (int v = 0; v < 8; ++v) {
                int m = m0 + waveM * 64 + tm * 16 + v + 8 * g;
                int n = n0 + waveN * 64 + tn * 16 + r16;
                float val = acc[tm][tn][v];
                if constexpr (EPI)
                    val += dskip[n] * xs[(size_t)m * DIM + n];
                Out[(size_t)m * DIM + n] = val;
            }
}

// ----------------------------- chunked scan --------------------------------
__global__ __launch_bounds__(256) void scan_pass1(float* __restrict__ B,
                                                  const float* __restrict__ lam,
                                                  float* __restrict__ F) {
    int n = blockIdx.x * blockDim.x + threadIdx.x;
    int c = blockIdx.y;
    float l = lam[n];
    float h = 0.f;
    size_t base = (size_t)c * CHUNK * DIM + n;
    for (int i = 0; i < CHUNK; ++i) {
        h = fmaf(l, h, B[base + (size_t)i * DIM]);
        B[base + (size_t)i * DIM] = h;
    }
    F[(size_t)c * DIM + n] = h;
}

__global__ __launch_bounds__(256) void scan_pass2(const float* __restrict__ F,
                                                  const float* __restrict__ lam,
                                                  float* __restrict__ Cin) {
    int n = blockIdx.x * blockDim.x + threadIdx.x;
    float l = lam[n];
    float lt = l;                       // lam^CHUNK via 8 squarings (CHUNK=2^8)
    #pragma unroll
    for (int i = 0; i < 8; ++i) lt *= lt;
    float carry = 0.f;
    for (int c = 0; c < NCHUNK; ++c) {
        Cin[(size_t)c * DIM + n] = carry;
        carry = fmaf(lt, carry, F[(size_t)c * DIM + n]);
    }
}

__global__ __launch_bounds__(256) void scan_pass3(const float* __restrict__ B,
                                                  const float* __restrict__ lam,
                                                  const float* __restrict__ Cin,
                                                  __bf16* __restrict__ H) {
    int n = blockIdx.x * blockDim.x + threadIdx.x;
    int c = blockIdx.y;
    float l = lam[n];
    float carry = Cin[(size_t)c * DIM + n];
    float p = l;
    size_t base = (size_t)c * CHUNK * DIM + n;
    for (int i = 0; i < CHUNK; ++i) {
        float h = fmaf(p, carry, B[base + (size_t)i * DIM]);
        p *= l;
        H[base + (size_t)i * DIM] = (__bf16)h;
    }
}

// ------------------------------- launcher ----------------------------------
extern "C" void kernel_launch(void* const* d_in, const int* in_sizes, int n_in,
                              void* d_out, int out_size, void* d_ws, size_t ws_size,
                              hipStream_t stream) {
    const float* xs    = (const float*)d_in[0];
    const float* lam   = (const float*)d_in[1];
    const float* w_in  = (const float*)d_in[2];
    const float* c_out = (const float*)d_in[3];
    const float* dskip = (const float*)d_in[4];
    float* out = (float*)d_out;

    char* ws = (char*)d_ws;
    const size_t SZ_W  = (size_t)DIM * DIM * sizeof(__bf16);       // 8 MiB
    const size_t SZ_XB = (size_t)SEQ_LEN * DIM * sizeof(__bf16);   // 64 MiB
    const size_t SZ_B  = (size_t)SEQ_LEN * DIM * sizeof(float);    // 128 MiB
    const size_t SZ_F  = (size_t)NCHUNK * DIM * sizeof(float);     // 512 KiB
    __bf16* w_bf = (__bf16*)(ws);
    __bf16* c_bf = (__bf16*)(ws + SZ_W);
    __bf16* x_bf = (__bf16*)(ws + 2 * SZ_W);
    float*  Bmat = (float*) (ws + 2 * SZ_W + SZ_XB);
    __bf16* H_bf = (__bf16*)(ws + 2 * SZ_W + SZ_XB + SZ_B);
    float*  Fch  = (float*) (ws + 2 * SZ_W + SZ_XB + SZ_B + SZ_XB);
    float*  Cin  = (float*) (ws + 2 * SZ_W + SZ_XB + SZ_B + SZ_XB + SZ_F);

    // 1) bf16 conversions (weights + activations)
    {
        size_t n4w = (size_t)DIM * DIM / 4;
        int bw = (int)((n4w + 255) / 256);
        cvt_f32_to_bf16<<<bw, 256, 0, stream>>>(w_in, w_bf, n4w);
        cvt_f32_to_bf16<<<bw, 256, 0, stream>>>(c_out, c_bf, n4w);
        size_t n4x = (size_t)SEQ_LEN * DIM / 4;
        int bx = (int)((n4x + 255) / 256);
        cvt_f32_to_bf16<<<bx, 256, 0, stream>>>(xs, x_bf, n4x);
    }

    dim3 ggrid(DIM / BN, SEQ_LEN / BM);   // (16, 64)

    // 2) B = xs @ w_in^T
    gemm_wmma_bf16<false><<<ggrid, 256, 0, stream>>>(x_bf, w_bf, Bmat, nullptr, nullptr);

    // 3) chunked associative scan
    dim3 sgrid(DIM / 256, NCHUNK);        // (8, 64)
    scan_pass1<<<sgrid, 256, 0, stream>>>(Bmat, lam, Fch);
    scan_pass2<<<dim3(DIM / 256), 256, 0, stream>>>(Fch, lam, Cin);
    scan_pass3<<<sgrid, 256, 0, stream>>>(Bmat, lam, Cin, H_bf);

    // 4) Y = H @ c_out^T + d*xs
    gemm_wmma_bf16<true><<<ggrid, 256, 0, stream>>>(H_bf, c_bf, out, xs, dskip);
}
